// LSTMModel_50500225466868
// MI455X (gfx1250) — matrix operand order repaired and checked by reading, not compiled, not run on hardware
//
#include <hip/hip_runtime.h>

// Problem constants (match reference)
#define B_   64
#define T_   2048
#define DIN  128
#define H_   256
#define D4   1024          // 4*H
#define KTOT 384           // DIN + H  (fused [x_t | h] K dimension)
#define NWG  16            // persistent workgroups (each owns H/NWG = 16 h-cols)
#define BLK  256           // 8 waves of 32
#define HPAD 264           // padded LDS stride (bf16 elems) for h staging: 528 B

typedef __bf16 bf16_t;
typedef __attribute__((ext_vector_type(16))) __bf16 v16bf;
typedef __attribute__((ext_vector_type(8)))  float  v8f;

// Optional CDNA5 async global->LDS path (probe the toolchain, fall back safely).
#if defined(__has_builtin)
#  if __has_builtin(__builtin_amdgcn_global_load_async_to_lds_b128) && \
      __has_builtin(__builtin_amdgcn_s_wait_asynccnt)
#    define USE_ASYNC_LDS 1
#  endif
#endif

__device__ __forceinline__ float sigf(float x) { return 1.0f / (1.0f + __expf(-x)); }

// A-fragment (16x32 bf16, MxK): per-lane elements [k0..k0+7] and [k0+16..k0+23]
// (k0 already includes khalf*8) -> two b128 loads with a 16-element gap.
__device__ __forceinline__ v16bf load_frag_a(const bf16_t* p) {
  union { uint4 u[2]; v16bf v; } u;
  u.u[0] = *reinterpret_cast<const uint4*>(p);
  u.u[1] = *reinterpret_cast<const uint4*>(p + 16);
  return u.v;
}

// B-fragment (32x16 bf16, KxN): per-lane K values [k0..k0+7] and [k0+16..k0+23] of one column.
__device__ __forceinline__ v16bf load_frag_b(const bf16_t* p) {
  union { uint4 u[2]; v16bf v; } u;
  u.u[0] = *reinterpret_cast<const uint4*>(p);
  u.u[1] = *reinterpret_cast<const uint4*>(p + 16);
  return u.v;
}

#if defined(USE_ASYNC_LDS)
// Builtin signature (from hipcc diagnostic): param0 is AS1 pointer to v4i.
typedef __attribute__((vector_size(16))) int v4i_t;
typedef __attribute__((address_space(1))) v4i_t* as1_v4i_ptr;
typedef __attribute__((address_space(3))) v4i_t* as3_v4i_ptr;
__device__ __forceinline__ void async_copy_b128(const void* gsrc, void* ldst) {
  __builtin_amdgcn_global_load_async_to_lds_b128(
      (as1_v4i_ptr)(gsrc), (as3_v4i_ptr)(ldst), 0, 0);
}
#endif

// ---------------------------------------------------------------------------
// Prep kernels: convert weights/activations to bf16 in WMMA-friendly layouts.
// ---------------------------------------------------------------------------

// Wc[n][k] = (k < DIN) ? W_x[k][n] : W_h[k-DIN][n]  (per-column contiguous K).
__global__ void lstm_prep_weights(const float* __restrict__ Wx,
                                  const float* __restrict__ Wh,
                                  bf16_t* __restrict__ Wc) {
  const int n = blockIdx.x;      // 0..D4-1
  const int k = threadIdx.x;     // 0..KTOT-1
  const float v = (k < DIN) ? Wx[(size_t)k * D4 + n] : Wh[(size_t)(k - DIN) * D4 + n];
  Wc[(size_t)n * KTOT + k] = (bf16_t)v;
}

// xT[t][b][d] = bf16(x[b][t][d])  -- time-major contiguous per-step slice.
__global__ void lstm_prep_x(const float* __restrict__ x, bf16_t* __restrict__ xT) {
  const size_t idx = (size_t)blockIdx.x * blockDim.x + threadIdx.x;  // < T*B*DIN
  const int t  = (int)(idx / (B_ * DIN));
  const int r  = (int)(idx % (B_ * DIN));
  const int bb = r / DIN;
  const int d  = r % DIN;
  xT[idx] = (bf16_t)x[((size_t)bb * T_ + t) * DIN + d];
}

__global__ void lstm_init_state(bf16_t* __restrict__ hbuf,
                                unsigned int* __restrict__ counter) {
  const int i = blockIdx.x * blockDim.x + threadIdx.x;
  if (i < 2 * B_ * H_) hbuf[i] = (bf16_t)0.0f;   // h(0) = 0 (both ping-pong bufs)
  if (i == 0) *counter = 0u;                     // reset grid-sync counter
}

// ---------------------------------------------------------------------------
// Persistent LSTM scan: 16 WGs x 256 threads, one grid-sync per timestep.
// gates[64x1024] = [x_t | h] (64x384 bf16) @ Wc (384x1024 bf16), f32 acc.
// Latency plan per step: one L2 burst (x-frags -> regs, h -> LDS async), then
// all WMMA operands come from registers (B, x) or LDS (h).
// ---------------------------------------------------------------------------
__global__ void __launch_bounds__(BLK) lstm_scan_kernel(
    const bf16_t* __restrict__ xT,       // [T][B][DIN] bf16
    const bf16_t* __restrict__ Wc,       // [D4][KTOT]  bf16
    const float*  __restrict__ bias,     // [D4]        f32
    bf16_t*       __restrict__ hbuf,     // [2][B][H]   bf16 ping-pong
    float*        __restrict__ hfinal,   // [B][H]      f32 (written at t=T-1)
    unsigned int* __restrict__ counter)  // grid sync
{
  __shared__ __align__(16) bf16_t Hs[B_ * HPAD];  // 33 KB staged h (padded rows)
  __shared__ float Gates[B_][4][16];              // 16 KB
  __shared__ float Cs[B_ * 16];                   //  4 KB cell state
  __shared__ float BiasS[64];

  const int tid   = threadIdx.x;
  const int wg    = blockIdx.x;        // 0..NWG-1
  const int j0    = wg * 16;           // first owned h column
  const int w     = tid >> 5;          // wave id (wave32)
  const int lane  = tid & 31;
  const int g     = w & 3;             // gate group handled by this wave
  const int m0    = (w >> 2) * 2;      // two M tiles per wave
  const int m1    = m0 + 1;
  const int colq  = lane & 15;
  const int khalf = lane >> 4;

  if (tid < 64) BiasS[tid] = bias[(tid >> 4) * H_ + j0 + (tid & 15)];
  for (int e = tid; e < B_ * 16; e += BLK) Cs[e] = 0.0f;

  // Resident B fragments: one gate-group column slice, all 12 K-chunks.
  v16bf Bf[12];
  {
    const int ncol = g * H_ + j0 + colq;
    const bf16_t* wp = Wc + (size_t)ncol * KTOT;
#pragma unroll
    for (int kc = 0; kc < 12; ++kc)
      Bf[kc] = load_frag_b(wp + kc * 32 + khalf * 8);
  }
  __syncthreads();

#pragma unroll 1
  for (int t = 0; t < T_; ++t) {
    const bf16_t* hcur = hbuf + (size_t)(t & 1) * (B_ * H_);
    bf16_t*       hnxt = hbuf + (size_t)((t + 1) & 1) * (B_ * H_);
    const bf16_t* xt   = xT + (size_t)t * (B_ * DIN);

    // ---- (1) issue x A-fragment loads for this step (one clause, no wait yet)
    v16bf xf0[4], xf1[4];
#pragma unroll
    for (int kc = 0; kc < 4; ++kc) {
      const int kb = kc * 32 + khalf * 8;
      xf0[kc] = load_frag_a(xt + (m0 * 16 + colq) * DIN + kb);
      xf1[kc] = load_frag_a(xt + (m1 * 16 + colq) * DIN + kb);
    }

    // ---- (2) stage h (32 KB) into LDS as one burst: 8 x b128 per thread
#if defined(USE_ASYNC_LDS)
#pragma unroll
    for (int i = 0; i < 8; ++i) {
      const int ci  = tid + i * BLK;        // 16-byte chunk index (2048 total)
      const int row = ci >> 4, c16 = ci & 15;
      async_copy_b128(hcur + ci * 8, (char*)Hs + row * (HPAD * 2) + c16 * 16);
    }
#else
    uint4 htmp[8];
#pragma unroll
    for (int i = 0; i < 8; ++i)
      htmp[i] = *reinterpret_cast<const uint4*>(hcur + (tid + i * BLK) * 8);
#pragma unroll
    for (int i = 0; i < 8; ++i) {
      const int ci  = tid + i * BLK;
      const int row = ci >> 4, c16 = ci & 15;
      *reinterpret_cast<uint4*>((char*)Hs + row * (HPAD * 2) + c16 * 16) = htmp[i];
    }
#endif

    // ---- (3) prefetch next step's x slice into cache (global_prefetch_b8)
    if (t + 1 < T_ && tid < 128)
      __builtin_prefetch(xt + B_ * DIN + tid * 64, 0, 1);

#if defined(USE_ASYNC_LDS)
    __builtin_amdgcn_s_wait_asynccnt(0);
#endif
    __syncthreads();

    // ---- (4) gates = [x_t | h] @ Wc : 4 chunks from regs + 8 chunks from LDS
    v8f acc0 = {}, acc1 = {};
#pragma unroll
    for (int kc = 0; kc < 4; ++kc) {
      acc0 = __builtin_amdgcn_wmma_f32_16x16x32_bf16(false, xf0[kc], false, Bf[kc],
                                                     (short)0, acc0, false, false);
      acc1 = __builtin_amdgcn_wmma_f32_16x16x32_bf16(false, xf1[kc], false, Bf[kc],
                                                     (short)0, acc1, false, false);
    }
#pragma unroll
    for (int kc = 0; kc < 8; ++kc) {
      const int kb = kc * 32 + khalf * 8;
      const v16bf a0 = load_frag_a(Hs + (m0 * 16 + colq) * HPAD + kb);
      const v16bf a1 = load_frag_a(Hs + (m1 * 16 + colq) * HPAD + kb);
      acc0 = __builtin_amdgcn_wmma_f32_16x16x32_bf16(false, a0, false, Bf[4 + kc],
                                                     (short)0, acc0, false, false);
      acc1 = __builtin_amdgcn_wmma_f32_16x16x32_bf16(false, a1, false, Bf[4 + kc],
                                                     (short)0, acc1, false, false);
    }

    // ---- (5) accumulators -> LDS gate buffer (C layout: VGPR r -> M=r|r+8)
    {
      const int rb = khalf * 8;
#pragma unroll
      for (int r = 0; r < 8; ++r) {
        Gates[m0 * 16 + rb + r][g][colq] = acc0[r];
        Gates[m1 * 16 + rb + r][g][colq] = acc1[r];
      }
    }
    __syncthreads();

    // ---- (6) elementwise LSTM cell update for our 64x16 slice
    for (int e = tid; e < B_ * 16; e += BLK) {
      const int row = e >> 4;
      const int col = e & 15;
      const float iv = sigf(Gates[row][0][col] + BiasS[col]);
      const float fv = sigf(Gates[row][1][col] + BiasS[16 + col]);
      const float gv = tanhf(Gates[row][2][col] + BiasS[32 + col]);
      const float ov = sigf(Gates[row][3][col] + BiasS[48 + col]);
      const float c  = fv * Cs[e] + iv * gv;
      Cs[e] = c;
      const float hv = ov * tanhf(c);
      hnxt[row * H_ + j0 + col] = (bf16_t)hv;
      if (t == T_ - 1) hfinal[row * H_ + j0 + col] = hv;
    }

    // ---- (7) device-wide step barrier (monotonic counter, release/acquire)
    __threadfence();
    __syncthreads();
    if (tid == 0) {
      __hip_atomic_fetch_add(counter, 1u, __ATOMIC_RELEASE, __HIP_MEMORY_SCOPE_AGENT);
      const unsigned int target = (unsigned int)(NWG * (t + 1));
      while (__hip_atomic_load(counter, __ATOMIC_ACQUIRE, __HIP_MEMORY_SCOPE_AGENT) < target)
        __builtin_amdgcn_s_sleep(1);
    }
    __syncthreads();
  }
}

// ---------------------------------------------------------------------------
// Final FC: out[b][n] = h_final[b][:] . fc_w[:][n] + fc_b[n]   (64x128, tiny)
// ---------------------------------------------------------------------------
__global__ void lstm_fc_kernel(const float* __restrict__ hfin,
                               const float* __restrict__ fcw,
                               const float* __restrict__ fcb,
                               float* __restrict__ out) {
  const int idx = blockIdx.x * blockDim.x + threadIdx.x;  // < B_*128
  const int bb = idx >> 7;
  const int n  = idx & 127;
  float s = fcb[n];
#pragma unroll 4
  for (int k = 0; k < H_; ++k)
    s += hfin[bb * H_ + k] * fcw[k * 128 + n];
  out[idx] = s;
}

// ---------------------------------------------------------------------------
extern "C" void kernel_launch(void* const* d_in, const int* in_sizes, int n_in,
                              void* d_out, int out_size, void* d_ws, size_t ws_size,
                              hipStream_t stream) {
  (void)in_sizes; (void)n_in; (void)out_size; (void)ws_size;

  const float* x    = (const float*)d_in[0];  // [B,T,DIN]
  const float* W_x  = (const float*)d_in[1];  // [DIN,4H]
  const float* W_h  = (const float*)d_in[2];  // [H,4H]
  const float* b    = (const float*)d_in[3];  // [4H]
  const float* fc_w = (const float*)d_in[4];  // [H,128]
  const float* fc_b = (const float*)d_in[5];  // [128]
  float* out = (float*)d_out;                 // [B,128]

  // Workspace layout (bytes, all 256-aligned):
  char* ws = (char*)d_ws;
  const size_t WC_BYTES = (size_t)D4 * KTOT * 2;        //   768 KB
  const size_t XT_BYTES = (size_t)T_ * B_ * DIN * 2;    //    32 MB
  const size_t HB_BYTES = (size_t)2 * B_ * H_ * 2;      //    64 KB
  const size_t HF_BYTES = (size_t)B_ * H_ * 4;          //    64 KB
  bf16_t*       Wc      = (bf16_t*)(ws);
  bf16_t*       xT      = (bf16_t*)(ws + WC_BYTES);
  bf16_t*       hbuf    = (bf16_t*)(ws + WC_BYTES + XT_BYTES);
  float*        hfinal  = (float*) (ws + WC_BYTES + XT_BYTES + HB_BYTES);
  unsigned int* counter = (unsigned int*)(ws + WC_BYTES + XT_BYTES + HB_BYTES + HF_BYTES);

  lstm_prep_weights<<<D4, KTOT, 0, stream>>>(W_x, W_h, Wc);
  lstm_prep_x<<<(T_ * B_ * DIN) / 256, 256, 0, stream>>>(x, xT);
  lstm_init_state<<<(2 * B_ * H_ + 255) / 256, 256, 0, stream>>>(hbuf, counter);
  lstm_scan_kernel<<<NWG, BLK, 0, stream>>>(xT, Wc, b, hbuf, hfinal, counter);
  lstm_fc_kernel<<<(B_ * 128) / 256, 256, 0, stream>>>(hfinal, fc_w, fc_b, out);
}
